// EpiEPMP_83983790506434
// MI455X (gfx1250) — matrix operand order repaired and checked by reading, not compiled, not run on hardware
//
#include <hip/hip_runtime.h>
#include <hip/hip_fp16.h>
#include <hip/hip_bf16.h>

// ---------------------------------------------------------------------------
// EGNN-heavy antibody/antigen GNN forward.
// Dominant op: EGNN edge-MLP layer 2, (n^2,258)x(258,16) GEMM done with
// v_wmma_f32_16x16x32_f16 (K padded 258->288 = 9x32). Layer 1 is factorized
// as Hi[i]+Hj[j]+dist*w_d+b1 (the dist column makes the 129-wide GEMM
// separable), and the coordinate branch is dead code in the reference.
// Round-2 fixes: double-buffered A-fragment LDS loads (waits become
// dscnt<=2 instead of 9x dscnt==0 per tile), register-only gate epilogue
// via shfl_xor butterfly (mS LDS round-trip removed).
// ---------------------------------------------------------------------------

typedef __attribute__((ext_vector_type(16))) _Float16 v16h;
typedef __attribute__((ext_vector_type(8)))  _Float16 v8h;
typedef __attribute__((ext_vector_type(2)))  _Float16 v2h;
typedef __attribute__((ext_vector_type(8)))  float    v8f;

#define NAB 384
#define NAG 768
#define NDN 1152
#define DD  64
#define FF  62
#define KPAD 288
#define KC   9

__device__ __forceinline__ float rcpf_(float x){ return __builtin_amdgcn_rcpf(x); }
__device__ __forceinline__ float rsqf_(float x){ return __builtin_amdgcn_rsqf(x); }
__device__ __forceinline__ float sigmf_(float x){ return rcpf_(1.f + __expf(-x)); }
__device__ __forceinline__ float siluf_(float x){ return x*sigmf_(x); }
__device__ __forceinline__ float lreluf_(float x){ return x>0.f ? x : 0.2f*x; }
__device__ __forceinline__ float bneval_(float x, const float* bn, int f, int d){
  // layout (sorted keys): beta@0, gamma@d, mean@2d, var@3d
  return (x - bn[2*d+f]) * rsqf_(bn[3*d+f] + 1e-5f) * bn[d+f] + bn[f];
}
__device__ __forceinline__ unsigned fenc_(float f){
  unsigned u = __float_as_uint(f);
  return (u & 0x80000000u) ? ~u : (u | 0x80000000u);
}
__device__ __forceinline__ float fdec_(unsigned u){
  return (u & 0x80000000u) ? __uint_as_float(u & 0x7fffffffu) : __uint_as_float(~u);
}

// ------------------------------- generic GEMM ------------------------------
__global__ __launch_bounds__(256) void k_gemm_xw(const float* __restrict__ X,
    const float* __restrict__ W, float* __restrict__ H, int n, int F, int D){
  int idx = blockIdx.x*blockDim.x + threadIdx.x;
  if (idx >= n*D) return;
  int r = idx / D, c = idx % D;
  float s = 0.f;
  for (int f = 0; f < F; f++) s += X[r*F+f]*W[f*D+c];
  H[idx] = s;
}

// --------------------------------- GCN -------------------------------------
__global__ __launch_bounds__(256) void k_deg(const int* __restrict__ ei,
    float* __restrict__ deg, int E, int n){
  int e = blockIdx.x*blockDim.x + threadIdx.x;
  if (e >= E + n) return;
  int dst = (e < E) ? ei[E + e] : (e - E);
  atomicAdd(&deg[dst], 1.f);
}
__global__ __launch_bounds__(256) void k_dinv(float* __restrict__ deg, int n){
  int v = blockIdx.x*blockDim.x + threadIdx.x;
  if (v < n) deg[v] = rsqf_(deg[v]);
}
__global__ __launch_bounds__(256) void k_gcn_agg(const int* __restrict__ ei,
    const float* __restrict__ h, const float* __restrict__ dinv,
    float* __restrict__ agg, int E){
  int idx = blockIdx.x*blockDim.x + threadIdx.x;
  if (idx >= E*DD) return;
  int e = idx >> 6, f = idx & 63;
  int src = ei[e], dst = ei[E + e];
  atomicAdd(&agg[dst*DD + f], h[src*DD + f]*dinv[src]*dinv[dst]);
}
__global__ __launch_bounds__(256) void k_gcn_self(const float* __restrict__ h,
    const float* __restrict__ dinv, float* __restrict__ agg, int n){
  int idx = blockIdx.x*blockDim.x + threadIdx.x;
  if (idx >= n*DD) return;
  int v = idx >> 6;
  agg[idx] += h[idx]*dinv[v]*dinv[v];
}
__global__ __launch_bounds__(256) void k_bias_bn_relu(const float* __restrict__ agg,
    const float* __restrict__ b, const float* __restrict__ bn,
    float* __restrict__ out, int n, int D){
  int idx = blockIdx.x*blockDim.x + threadIdx.x;
  if (idx >= n*D) return;
  int f = idx % D;
  out[idx] = fmaxf(bneval_(agg[idx] + b[f], bn, f, D), 0.f);
}

// ------------------------------ EGNN precompute ----------------------------
// Hi = feats @ e_w1[0:64,:] + e_b1 ; Hj = feats @ e_w1[64:128,:] ; zero-padded
__global__ __launch_bounds__(256) void k_hij(const float* __restrict__ h0,
    const float* __restrict__ ew1, const float* __restrict__ eb1,
    float* __restrict__ Hi, float* __restrict__ Hj, int n){
  int idx = blockIdx.x*blockDim.x + threadIdx.x;
  if (idx >= n*KPAD) return;
  int v = idx / KPAD, k = idx % KPAD;
  float si = 0.f, sj = 0.f;
  if (k < 258) {
    si = eb1[k];
    const float* hv = h0 + v*DD;
    for (int f = 0; f < DD; f++) {
      si += hv[f]*ew1[f*258 + k];
      sj += hv[f]*ew1[(64 + f)*258 + k];
    }
  }
  Hi[idx] = si; Hj[idx] = sj;
}
__global__ __launch_bounds__(256) void k_wd(const float* __restrict__ ew1,
    float* __restrict__ wd){
  int k = blockIdx.x*blockDim.x + threadIdx.x;
  if (k < KPAD) wd[k] = (k < 258) ? ew1[128*258 + k] : 0.f;
}
// B fragment pack: Bf[(kc*32+lane)*16 + h] = e_w2[k,n], k = kc*32+(h/8)*16+(lane/16)*8+h%8, n = lane%16
__global__ __launch_bounds__(256) void k_bpack(const float* __restrict__ ew2,
    _Float16* __restrict__ Bf){
  int idx = blockIdx.x*blockDim.x + threadIdx.x;
  if (idx >= KC*32*16) return;
  int kc = idx / 512, lane = (idx / 16) & 31, hh = idx & 15;
  int k = kc*32 + (hh >> 3)*16 + (lane >> 4)*8 + (hh & 7);
  int nn = lane & 15;
  Bf[idx] = (k < 258) ? (_Float16)ew2[k*16 + nn] : (_Float16)0.f;
}

// ------------------------------ EGNN WMMA core -----------------------------
// 4 waves/block, one wave per node i; barrier-free.  Per j-tile of 16 rows:
//   A[r][k] = silu(Hi[i][k] + Hj[j0+r][k] + dist(i,j0+r)*wd[k])  (f16, LDS)
//   m_pre   = A(16x288) x e_w2(288x16)  via 9x v_wmma_f32_16x16x32_f16
//   m       = silu(m_pre + e_b2); gated by sigmoid(m.g_w + g_b); sum over j.
__global__ __launch_bounds__(128) void k_egnn(
    const float* __restrict__ Hi, const float* __restrict__ Hj,
    const float* __restrict__ coords, const float* __restrict__ wd288,
    const _Float16* __restrict__ Bf, const float* __restrict__ e_b2,
    const float* __restrict__ g_w, const float* __restrict__ g_b,
    float* __restrict__ Msum, int n)
{
  __shared__ _Float16 As[4][16*KPAD];

  const int tid  = threadIdx.x;
  const int wv   = tid >> 5;
  const int lane = tid & 31;
  const int sub  = lane >> 4;   // lane group: 0 or 1
  const int row  = lane & 15;
  const int i    = blockIdx.x*4 + wv;

  // B fragments straight into registers (hit L2/WGP$; reused for all tiles)
  v16h bfr[KC];
  #pragma unroll
  for (int kc = 0; kc < KC; kc++)
    bfr[kc] = *(const v16h*)(Bf + (kc*32 + lane)*16);

  // per-lane Hi / w_d pairs: k = 2*lane + 64*g (g<4), partial k in [256,288)
  float2 hp[4], wp[4];
  #pragma unroll
  for (int g = 0; g < 4; g++) {
    hp[g] = *(const float2*)(Hi + (size_t)i*KPAD + 2*lane + 64*g);
    wp[g] = *(const float2*)(wd288 + 2*lane + 64*g);
  }
  const int l2 = (lane & 15)*2;     // lanes 16-31 duplicate lanes 0-15
  float2 hp4 = *(const float2*)(Hi + (size_t)i*KPAD + 256 + l2);
  float2 wp4 = *(const float2*)(wd288 + 256 + l2);

  const float cix = coords[i*3+0], ciy = coords[i*3+1], ciz = coords[i*3+2];
  const float eb2n = e_b2[row];
  const float gwn  = g_w[row];
  const float gb   = g_b[0];

  _Float16* myA = As[wv];
  float macc = 0.f;

  for (int jc = 0; jc < n; jc += 16) {
    if (jc + 16 < n) __builtin_prefetch(Hj + (size_t)(jc+16)*KPAD, 0, 1);

    // squared distance for row (lane%16) of this tile
    int jj = jc + row;
    float dx = cix - coords[jj*3+0];
    float dy = ciy - coords[jj*3+1];
    float dz = ciz - coords[jj*3+2];
    float d2 = dx*dx + dy*dy + dz*dz;

    // build f16 A tile in LDS, two k's per lane per store (pads silu(0)=0)
    for (int r = 0; r < 16; r++) {
      float dr = __shfl(d2, r, 32);
      const float* hjr = Hj + (size_t)(jc + r)*KPAD;
      #pragma unroll
      for (int g = 0; g < 4; g++) {
        float2 hj = *(const float2*)(hjr + 2*lane + 64*g);
        v2h pk;
        pk[0] = (_Float16)siluf_(hp[g].x + hj.x + dr*wp[g].x);
        pk[1] = (_Float16)siluf_(hp[g].y + hj.y + dr*wp[g].y);
        *(v2h*)(myA + r*KPAD + 2*lane + 64*g) = pk;
      }
      {
        float2 hj = *(const float2*)(hjr + 256 + l2);
        v2h pk;
        pk[0] = (_Float16)siluf_(hp4.x + hj.x + dr*wp4.x);
        pk[1] = (_Float16)siluf_(hp4.y + hj.y + dr*wp4.y);
        *(v2h*)(myA + r*KPAD + 256 + l2) = pk;
      }
    }

    // double-buffered fragment loads: loads for kc+1 issue before the wmma
    // consuming kc, so each wait is dscnt<=2 (DS returns in order), not 0.
    const _Float16* Ab = myA + row*KPAD;
    v8h lo  = *(const v8h*)(Ab + sub*8);
    v8h hi2 = *(const v8h*)(Ab + 16 + sub*8);
    v8f c = {};
    #pragma unroll
    for (int kc = 0; kc < KC; kc++) {
      v8h lo_n = lo, hi_n = hi2;
      if (kc + 1 < KC) {
        lo_n = *(const v8h*)(Ab + (kc+1)*32 + sub*8);
        hi_n = *(const v8h*)(Ab + (kc+1)*32 + 16 + sub*8);
      }
      v16h a;
      #pragma unroll
      for (int t = 0; t < 8; t++) { a[t] = lo[t]; a[t+8] = hi2[t]; }
      c = __builtin_amdgcn_wmma_f32_16x16x32_f16(false, a, false, bfr[kc],
                                                 (short)0, c, false, false);
      lo = lo_n; hi2 = hi_n;
    }

    // register-only epilogue.  C layout: c[v] = m_pre[M=v+8*sub][N=row].
    // m[v] = silu(c[v]+b2[N]);  gate dot over N via 16-lane xor butterfly.
    float mreg[8], t[8];
    #pragma unroll
    for (int v = 0; v < 8; v++) {
      mreg[v] = siluf_(c[v] + eb2n);
      t[v]    = mreg[v]*gwn;
    }
    #pragma unroll
    for (int mask = 1; mask <= 8; mask <<= 1) {
      #pragma unroll
      for (int v = 0; v < 8; v++) t[v] += __shfl_xor(t[v], mask, 32);
    }
    // t[v] now = sum_N m[row=v+8*sub][N]*g_w[N] for every lane in the half.
    #pragma unroll
    for (int v = 0; v < 8; v++)
      macc += mreg[v] * sigmf_(t[v] + gb);   // lane's feature n = row
  }
  macc += __shfl_down(macc, 16, 32);
  if (lane < 16) Msum[(size_t)i*16 + row] = macc;
}

// ------------------------------ EGNN node MLP ------------------------------
__global__ __launch_bounds__(256) void k_node1(const float* __restrict__ h0,
    const float* __restrict__ Msum, const float* __restrict__ nw1,
    const float* __restrict__ nb1, float* __restrict__ tmp, int n){
  int idx = blockIdx.x*blockDim.x + threadIdx.x;
  if (idx >= n*128) return;
  int v = idx >> 7, c = idx & 127;
  float s = nb1[c];
  const float* hv = h0 + v*DD;
  for (int f = 0; f < DD; f++) s += hv[f]*nw1[f*128 + c];
  const float* mv = Msum + v*16;
  for (int m = 0; m < 16; m++) s += mv[m]*nw1[(64 + m)*128 + c];
  tmp[idx] = siluf_(s);
}
__global__ __launch_bounds__(256) void k_node2(const float* __restrict__ tmp,
    const float* __restrict__ nw2, const float* __restrict__ nb2,
    const float* __restrict__ h0, const float* __restrict__ bn,
    float* __restrict__ out, int n){
  int idx = blockIdx.x*blockDim.x + threadIdx.x;
  if (idx >= n*DD) return;
  int v = idx >> 6, f = idx & 63;
  float s = nb2[f] + h0[idx];
  const float* tv = tmp + v*128;
  for (int c = 0; c < 128; c++) s += tv[c]*nw2[c*DD + f];
  out[idx] = fmaxf(bneval_(s, bn, f, DD), 0.f);
}

// --------------------------------- GAT -------------------------------------
__global__ __launch_bounds__(256) void k_gat_sd(const float* __restrict__ gh,
    const float* __restrict__ a_src, const float* __restrict__ a_dst,
    float* __restrict__ S, float* __restrict__ Dv, int n){
  int v = blockIdx.x*blockDim.x + threadIdx.x;
  if (v >= n) return;
  float s = 0.f, d = 0.f;
  const float* hv = gh + v*DD;
  for (int f = 0; f < DD; f++) { s += hv[f]*a_src[f]; d += hv[f]*a_dst[f]; }
  S[v] = s; Dv[v] = d;
}
__global__ __launch_bounds__(256) void k_gat_max(const int* __restrict__ ei,
    const float* __restrict__ S, const float* __restrict__ Dv,
    unsigned* __restrict__ mxu, int E, int n){
  int e = blockIdx.x*blockDim.x + threadIdx.x;
  if (e >= E + n) return;
  int src = (e < E) ? ei[e] : (e - E);
  int dst = (e < E) ? ei[E + e] : (e - E);
  atomicMax(&mxu[dst], fenc_(lreluf_(S[src] + Dv[dst])));
}
__global__ __launch_bounds__(256) void k_gat_den(const int* __restrict__ ei,
    const float* __restrict__ S, const float* __restrict__ Dv,
    const unsigned* __restrict__ mxu, float* __restrict__ den, int E, int n){
  int e = blockIdx.x*blockDim.x + threadIdx.x;
  if (e >= E + n) return;
  int src = (e < E) ? ei[e] : (e - E);
  int dst = (e < E) ? ei[E + e] : (e - E);
  atomicAdd(&den[dst], __expf(lreluf_(S[src] + Dv[dst]) - fdec_(mxu[dst])));
}
__global__ __launch_bounds__(256) void k_gat_agg(const int* __restrict__ ei,
    const float* __restrict__ gh, const float* __restrict__ S,
    const float* __restrict__ Dv, const unsigned* __restrict__ mxu,
    const float* __restrict__ den, float* __restrict__ out, int E, int n){
  int e = blockIdx.x*blockDim.x + threadIdx.x;
  if (e >= E + n) return;
  int src = (e < E) ? ei[e] : (e - E);
  int dst = (e < E) ? ei[E + e] : (e - E);
  float ex = __expf(lreluf_(S[src] + Dv[dst]) - fdec_(mxu[dst]));
  float alpha = ex * rcpf_(fmaxf(den[dst], 1e-16f));
  const float* hs = gh + src*DD;
  float* od = out + dst*DD;
  for (int f = 0; f < DD; f++) atomicAdd(&od[f], hs[f]*alpha);
}
__global__ __launch_bounds__(256) void k_gat_fin(float* __restrict__ out,
    const float* __restrict__ bias, int do_relu, int n){
  int idx = blockIdx.x*blockDim.x + threadIdx.x;
  if (idx >= n*DD) return;
  float x = out[idx] + bias[idx & 63];
  out[idx] = do_relu ? fmaxf(x, 0.f) : x;
}

// --------------------------------- heads -----------------------------------
__global__ __launch_bounds__(256) void k_final(const float* __restrict__ gout2,
    const float* __restrict__ h1ab, const float* __restrict__ h1ag,
    const float* __restrict__ bn2, const float* __restrict__ agbn2,
    const float* __restrict__ fcw, const float* __restrict__ fcb,
    const float* __restrict__ agfcw, const float* __restrict__ agfcb,
    float* __restrict__ out){
  int idx = blockIdx.x*blockDim.x + threadIdx.x;
  if (idx >= NDN) return;
  const float* g  = gout2 + idx*DD;
  const float* h  = (idx < NAB) ? (h1ab + idx*DD) : (h1ag + (idx - NAB)*DD);
  const float* bn = (idx < NAB) ? bn2 : agbn2;
  const float* w  = (idx < NAB) ? fcw : agfcw;
  float s = (idx < NAB) ? fcb[0] : agfcb[0];
  for (int f = 0; f < 128; f++) {
    float x = (f < DD) ? g[f] : h[f - DD];
    s += fmaxf(bneval_(x, bn, f, 128), 0.f)*w[f];
  }
  out[idx] = s;
}

// ---------------------------------------------------------------------------
static inline int LB(int n){ return (n + 255) / 256; }

extern "C" void kernel_launch(void* const* d_in, const int* in_sizes, int n_in,
                              void* d_out, int out_size, void* d_ws, size_t ws_size,
                              hipStream_t stream) {
  // Resolve inputs by flat size (all distinct); fallback to positional order.
  const float *x_ab=nullptr,*x_ag=nullptr,*cab=nullptr,*cag=nullptr,*P=nullptr;
  const int *exab=nullptr,*exag=nullptr,*eid=nullptr;
  for (int i = 0; i < n_in; i++) {
    switch (in_sizes[i]) {
      case NAB*FF:   x_ab=(const float*)d_in[i]; break;
      case NAG*FF:   x_ag=(const float*)d_in[i]; break;
      case 2*16*NAB: exab=(const int*)d_in[i];   break;
      case 2*16*NAG: exag=(const int*)d_in[i];   break;
      case 2*32*NDN: eid =(const int*)d_in[i];   break;
      case NAB*3:    cab =(const float*)d_in[i]; break;
      case NAG*3:    cag =(const float*)d_in[i]; break;
      default: if (in_sizes[i] >= 133904) P=(const float*)d_in[i]; break;
    }
  }
  if (!P && n_in >= 8) { // positional fallback (setup_inputs insertion order)
    x_ab=(const float*)d_in[0]; x_ag=(const float*)d_in[1];
    exab=(const int*)d_in[2];   exag=(const int*)d_in[3];
    eid =(const int*)d_in[4];
    cab =(const float*)d_in[5]; cag=(const float*)d_in[6];
    P   =(const float*)d_in[7];
  }

  // params: pytree leaves, sorted dict keys / tuples in order, concatenated
  const float* abbn0 = P + 0;
  const float* agbn0 = P + 256;
  const float* agbn2 = P + 512;
  const float* agfcw = P + 1024;   const float* agfcb = P + 1024 + 128;
  const float* bn1ab = P + 1153;
  const float* bn1ag = P + 1409;
  const float* bn2   = P + 1665;
  const float* egab  = P + 2177;
  const float* egag  = P + 59656;
  const float* fcw   = P + 117135; const float* fcb   = P + 117135 + 128;
  const float* gat1  = P + 117264;
  const float* gat2  = P + 121552;
  const float* gcabw = P + 125840; const float* gcabb = P + 125840 + 3968;
  const float* gcagw = P + 129872; const float* gcagb = P + 129872 + 3968;
  // egnn sub-offsets (sorted keys; coord branch unused)
  const int EG_EB1=1154, EG_EB2=1412, EG_EW1=1428, EG_EW2=34710;
  const int EG_GB=38838, EG_GW=38839, EG_NB1=38855, EG_NB2=38983;
  const int EG_NW1=39047, EG_NW2=49287;
  // gat sub-offsets: W@0, a_dst@4096, a_src@4160, bias@4224

  // workspace carve
  char* wsb = (char*)d_ws; size_t off = 0;
  auto alloc = [&](size_t bytes)->void* {
    void* p = wsb + off; off = (off + bytes + 255) & ~(size_t)255; return p;
  };
  float* dinv_ab=(float*)alloc(NAB*4);        float* dinv_ag=(float*)alloc(NAG*4);
  float* gh_ab =(float*)alloc(NAB*DD*4);      float* gh_ag =(float*)alloc(NAG*DD*4);
  float* agg_ab=(float*)alloc(NAB*DD*4);      float* agg_ag=(float*)alloc(NAG*DD*4);
  float* h0_ab =(float*)alloc(NAB*DD*4);      float* h0_ag =(float*)alloc(NAG*DD*4);
  float* Hi_ab =(float*)alloc((size_t)NAB*KPAD*4); float* Hj_ab=(float*)alloc((size_t)NAB*KPAD*4);
  float* Hi_ag =(float*)alloc((size_t)NAG*KPAD*4); float* Hj_ag=(float*)alloc((size_t)NAG*KPAD*4);
  float* wd_ab =(float*)alloc(KPAD*4);        float* wd_ag =(float*)alloc(KPAD*4);
  _Float16* Bf_ab=(_Float16*)alloc(KC*32*16*2); _Float16* Bf_ag=(_Float16*)alloc(KC*32*16*2);
  float* Ms_ab =(float*)alloc(NAB*16*4);      float* Ms_ag =(float*)alloc(NAG*16*4);
  float* tmp_ab=(float*)alloc(NAB*128*4);     float* tmp_ag=(float*)alloc(NAG*128*4);
  float* xcat  =(float*)alloc((size_t)NDN*DD*4);  // h1_ab | h1_ag
  float* h1_ab = xcat; float* h1_ag = xcat + (size_t)NAB*DD;
  float* ghat  =(float*)alloc((size_t)NDN*DD*4);
  float* gS    =(float*)alloc(NDN*4);  float* gD =(float*)alloc(NDN*4);
  unsigned* mxu=(unsigned*)alloc(NDN*4); float* den=(float*)alloc(NDN*4);
  float* gout1 =(float*)alloc((size_t)NDN*DD*4);
  float* ghat2 =(float*)alloc((size_t)NDN*DD*4);
  float* gS2   =(float*)alloc(NDN*4);  float* gD2=(float*)alloc(NDN*4);
  unsigned* mxu2=(unsigned*)alloc(NDN*4); float* den2=(float*)alloc(NDN*4);
  float* gout2 =(float*)alloc((size_t)NDN*DD*4);

  hipMemsetAsync(d_ws, 0, off, stream);  // zeros accumulators; 0u < fenc(any finite)

  const int EAB = 16*NAB, EAG = 16*NAG, ED = 32*NDN;

  // ---- GCN (ab / ag) ----
  k_gemm_xw<<<LB(NAB*DD),256,0,stream>>>(x_ab, gcabw, gh_ab, NAB, FF, DD);
  k_gemm_xw<<<LB(NAG*DD),256,0,stream>>>(x_ag, gcagw, gh_ag, NAG, FF, DD);
  k_deg<<<LB(EAB+NAB),256,0,stream>>>(exab, dinv_ab, EAB, NAB);
  k_deg<<<LB(EAG+NAG),256,0,stream>>>(exag, dinv_ag, EAG, NAG);
  k_dinv<<<LB(NAB),256,0,stream>>>(dinv_ab, NAB);
  k_dinv<<<LB(NAG),256,0,stream>>>(dinv_ag, NAG);
  k_gcn_agg<<<LB(EAB*DD),256,0,stream>>>(exab, gh_ab, dinv_ab, agg_ab, EAB);
  k_gcn_agg<<<LB(EAG*DD),256,0,stream>>>(exag, gh_ag, dinv_ag, agg_ag, EAG);
  k_gcn_self<<<LB(NAB*DD),256,0,stream>>>(gh_ab, dinv_ab, agg_ab, NAB);
  k_gcn_self<<<LB(NAG*DD),256,0,stream>>>(gh_ag, dinv_ag, agg_ag, NAG);
  k_bias_bn_relu<<<LB(NAB*DD),256,0,stream>>>(agg_ab, gcabb, abbn0, h0_ab, NAB, DD);
  k_bias_bn_relu<<<LB(NAG*DD),256,0,stream>>>(agg_ag, gcagb, agbn0, h0_ag, NAG, DD);

  // ---- EGNN ab ----
  k_hij<<<LB(NAB*KPAD),256,0,stream>>>(h0_ab, egab+EG_EW1, egab+EG_EB1, Hi_ab, Hj_ab, NAB);
  k_wd<<<LB(KPAD),256,0,stream>>>(egab+EG_EW1, wd_ab);
  k_bpack<<<LB(KC*32*16),256,0,stream>>>(egab+EG_EW2, Bf_ab);
  k_egnn<<<NAB/4,128,0,stream>>>(Hi_ab, Hj_ab, cab, wd_ab, Bf_ab,
                                 egab+EG_EB2, egab+EG_GW, egab+EG_GB, Ms_ab, NAB);
  k_node1<<<LB(NAB*128),256,0,stream>>>(h0_ab, Ms_ab, egab+EG_NW1, egab+EG_NB1, tmp_ab, NAB);
  k_node2<<<LB(NAB*DD),256,0,stream>>>(tmp_ab, egab+EG_NW2, egab+EG_NB2, h0_ab, bn1ab, h1_ab, NAB);

  // ---- EGNN ag ----
  k_hij<<<LB(NAG*KPAD),256,0,stream>>>(h0_ag, egag+EG_EW1, egag+EG_EB1, Hi_ag, Hj_ag, NAG);
  k_wd<<<LB(KPAD),256,0,stream>>>(egag+EG_EW1, wd_ag);
  k_bpack<<<LB(KC*32*16),256,0,stream>>>(egag+EG_EW2, Bf_ag);
  k_egnn<<<NAG/4,128,0,stream>>>(Hi_ag, Hj_ag, cag, wd_ag, Bf_ag,
                                 egag+EG_EB2, egag+EG_GW, egag+EG_GB, Ms_ag, NAG);
  k_node1<<<LB(NAG*128),256,0,stream>>>(h0_ag, Ms_ag, egag+EG_NW1, egag+EG_NB1, tmp_ag, NAG);
  k_node2<<<LB(NAG*DD),256,0,stream>>>(tmp_ag, egag+EG_NW2, egag+EG_NB2, h0_ag, bn1ag, h1_ag, NAG);

  // ---- GAT layer 1 ----
  k_gemm_xw<<<LB(NDN*DD),256,0,stream>>>(xcat, gat1+0, ghat, NDN, DD, DD);
  k_gat_sd<<<LB(NDN),256,0,stream>>>(ghat, gat1+4160, gat1+4096, gS, gD, NDN);
  k_gat_max<<<LB(ED+NDN),256,0,stream>>>(eid, gS, gD, mxu, ED, NDN);
  k_gat_den<<<LB(ED+NDN),256,0,stream>>>(eid, gS, gD, mxu, den, ED, NDN);
  k_gat_agg<<<LB(ED+NDN),256,0,stream>>>(eid, ghat, gS, gD, mxu, den, gout1, ED, NDN);
  k_gat_fin<<<LB(NDN*DD),256,0,stream>>>(gout1, gat1+4224, 1, NDN);

  // ---- GAT layer 2 ----
  k_gemm_xw<<<LB(NDN*DD),256,0,stream>>>(gout1, gat2+0, ghat2, NDN, DD, DD);
  k_gat_sd<<<LB(NDN),256,0,stream>>>(ghat2, gat2+4160, gat2+4096, gS2, gD2, NDN);
  k_gat_max<<<LB(ED+NDN),256,0,stream>>>(eid, gS2, gD2, mxu2, ED, NDN);
  k_gat_den<<<LB(ED+NDN),256,0,stream>>>(eid, gS2, gD2, mxu2, den2, ED, NDN);
  k_gat_agg<<<LB(ED+NDN),256,0,stream>>>(eid, ghat2, gS2, gD2, mxu2, den2, gout2, ED, NDN);
  k_gat_fin<<<LB(NDN*DD),256,0,stream>>>(gout2, gat2+4224, 0, NDN);

  // ---- heads ----
  k_final<<<LB(NDN),256,0,stream>>>(gout2, h1_ab, h1_ag, bn2, agbn2,
                                    fcw, fcb, agfcw, agfcb, (float*)d_out);
}
